// MT0Attention_28209345200169
// MI455X (gfx1250) — matrix operand order repaired and checked by reading, not compiled, not run on hardware
//
#include <hip/hip_runtime.h>
#include <cstdint>
#include <cstddef>

// ---------------------------------------------------------------------------
// Problem constants (match reference)
// ---------------------------------------------------------------------------
#define S_LEN   2048
#define BATCH   2
#define NH      16
#define DKV     64
#define DMODEL  1024
#define MTOT    (BATCH * S_LEN)      // 4096 rows for the big GEMMs
#define NDELTA  (2 * S_LEN - 1)      // 4095 relative positions

// GEMM tiling
#define TK 32
#define LDSTR 40                     // padded LDS row stride in ushorts (80B: 16B aligned, bank-spread)

typedef __attribute__((ext_vector_type(16))) __bf16 v16bf;
typedef __attribute__((ext_vector_type(8)))  float  v8f;

union FragAB {
    uint4 u[2];
    v16bf v;
};

struct __align__(8) us4 { unsigned short x, y, z, w; };

__device__ __forceinline__ unsigned short f2bf(float f) {
    union { float f; unsigned u; } x; x.f = f;
    unsigned u = x.u;
    unsigned r = u + 0x7FFFu + ((u >> 16) & 1u);   // round-to-nearest-even
    return (unsigned short)(r >> 16);
}

__device__ __forceinline__ v8f v8f_zero() {
    v8f z;
#pragma unroll
    for (int i = 0; i < 8; ++i) z[i] = 0.0f;
    return z;
}

__device__ __forceinline__ v8f wmma_bf16(v16bf a, v16bf b, v8f c) {
    // D = A(16x32 bf16) * B(32x16 bf16) + C(16x16 f32)
    return __builtin_amdgcn_wmma_f32_16x16x32_bf16(
        /*neg_a=*/false, a, /*neg_b=*/false, b,
        /*c_mod=*/(short)0, c, /*reuse_a=*/false, /*reuse_b=*/false);
}

// ---------------------------------------------------------------------------
// Kernel 1: relative-position bias table  bias_tab[h][delta + 2047]
// ---------------------------------------------------------------------------
__global__ void bias_table_kernel(const float* __restrict__ rel_bias,
                                  float* __restrict__ bias_tab) {
    int idx = blockIdx.x * blockDim.x + threadIdx.x;
    if (idx >= NH * NDELTA) return;
    int h  = idx / NDELTA;
    int di = idx % NDELTA;
    int delta = di - (S_LEN - 1);                 // mem - ctx
    int rb = (delta > 0) ? 16 : 0;                // bidirectional: num_buckets//2 = 16
    int rp = (delta < 0) ? -delta : delta;
    int v;
    if (rp < 8) {                                 // max_exact = 8
        v = rp;
    } else {
        float t = __logf((float)rp * 0.125f) * (1.0f / __logf(16.0f)) * 8.0f;
        v = 8 + (int)t;
        if (v > 15) v = 15;
    }
    bias_tab[idx] = rel_bias[(rb + v) * NH + h];
}

// ---------------------------------------------------------------------------
// Kernel 2: fused QKV projection.
//   X[4096,1024] f32  x  {Wq,Wk,Wv}[1024,1024] f32
//   -> Qbf, Kbf as bf16 [B,H,S,DKV];  Vt as bf16 [B,H,DKV,S] (transposed)
// Block = 128 threads (4 waves). Block tile 64x64, wave tile 16x64 (x3 outputs).
// ---------------------------------------------------------------------------
__global__ __launch_bounds__(128)
void qkv_proj_kernel(const float* __restrict__ X,
                     const float* __restrict__ Wq,
                     const float* __restrict__ Wk,
                     const float* __restrict__ Wv,
                     unsigned short* __restrict__ Qbf,
                     unsigned short* __restrict__ Kbf,
                     unsigned short* __restrict__ Vt) {
    __shared__ unsigned short Alds[64 * LDSTR];          // [m][k] bf16
    __shared__ unsigned short Blds[3][64 * LDSTR];       // transposed: [n][k] bf16

    const int tid  = threadIdx.x;
    const int wave = tid >> 5;
    const int lane = tid & 31;
    const int hi   = lane >> 4;
    const int lcol = lane & 15;
    const int m0 = blockIdx.y * 64;
    const int n0 = blockIdx.x * 64;

    const float* Ws[3] = { Wq, Wk, Wv };

    v8f acc[3][4];
#pragma unroll
    for (int o = 0; o < 3; ++o)
#pragma unroll
        for (int t = 0; t < 4; ++t) acc[o][t] = v8f_zero();

    const int arow = tid >> 1, ahalf = tid & 1;   // A staging: 64 rows x 2 halves
    const int bk = tid >> 2, bq = tid & 3;        // B staging: 32 k-rows x 4 n-quarters

    for (int k0 = 0; k0 < DMODEL; k0 += TK) {
        // --- prefetch next K tile (global_prefetch_b8) ---
        if (k0 + TK < DMODEL) {
            __builtin_prefetch(X + (size_t)(m0 + arow) * DMODEL + (k0 + TK) + ahalf * 16, 0, 1);
            __builtin_prefetch(Ws[0] + (size_t)(k0 + TK + bk) * DMODEL + n0 + bq * 16, 0, 1);
        }

        // --- stage A tile (f32 -> bf16), row-major ---
        {
            const float4* src = (const float4*)(X + (size_t)(m0 + arow) * DMODEL + k0 + ahalf * 16);
            us4* dst = (us4*)(Alds + arow * LDSTR + ahalf * 16);
#pragma unroll
            for (int j = 0; j < 4; ++j) {
                float4 f = src[j];
                us4 h;
                h.x = f2bf(f.x); h.y = f2bf(f.y); h.z = f2bf(f.z); h.w = f2bf(f.w);
                dst[j] = h;
            }
        }
        // --- stage 3 B tiles transposed: Blds[o][n][k] ---
#pragma unroll
        for (int o = 0; o < 3; ++o) {
            const float4* src = (const float4*)(Ws[o] + (size_t)(k0 + bk) * DMODEL + n0 + bq * 16);
#pragma unroll
            for (int j = 0; j < 4; ++j) {
                float4 f = src[j];
                int nb = bq * 16 + j * 4;
                Blds[o][(nb + 0) * LDSTR + bk] = f2bf(f.x);
                Blds[o][(nb + 1) * LDSTR + bk] = f2bf(f.y);
                Blds[o][(nb + 2) * LDSTR + bk] = f2bf(f.z);
                Blds[o][(nb + 3) * LDSTR + bk] = f2bf(f.w);
            }
        }
        __syncthreads();

        // --- A fragment (16x32 bf16, wave32 A-layout) ---
        FragAB af;
        {
            const unsigned short* ap = Alds + (wave * 16 + lcol) * LDSTR + hi * 8;
            af.u[0] = *(const uint4*)(ap);        // K 0..7  (lo) / 8..15  (hi)
            af.u[1] = *(const uint4*)(ap + 16);   // K 16..23 (lo) / 24..31 (hi)
        }
        // --- B fragments batched into distinct registers, then WMMA batch:
        //     lets the compiler issue 8 ds_load_b128 back-to-back and overlap
        //     group o+1 loads with group o WMMAs (no per-WMMA s_wait_dscnt 0) ---
#pragma unroll
        for (int o = 0; o < 3; ++o) {
            FragAB bfg[4];
#pragma unroll
            for (int t = 0; t < 4; ++t) {
                const unsigned short* bp = &Blds[o][(t * 16 + lcol) * LDSTR + hi * 8];
                bfg[t].u[0] = *(const uint4*)(bp);
                bfg[t].u[1] = *(const uint4*)(bp + 16);
            }
#pragma unroll
            for (int t = 0; t < 4; ++t)
                acc[o][t] = wmma_bf16(af.v, bfg[t].v, acc[o][t]);
        }
        __syncthreads();
    }

    // --- epilogue: scatter into attention layouts ---
#pragma unroll
    for (int t = 0; t < 4; ++t) {
#pragma unroll
        for (int r = 0; r < 8; ++r) {
            int m = m0 + wave * 16 + r + hi * 8;
            int n = n0 + t * 16 + lcol;
            int b = m >> 11;        // / 2048
            int s = m & (S_LEN - 1);
            int h = n >> 6;         // / 64
            int dk = n & (DKV - 1);
            size_t qk_idx = (((size_t)(b * NH + h) * S_LEN) + s) * DKV + dk;
            Qbf[qk_idx] = f2bf(acc[0][t][r]);
            Kbf[qk_idx] = f2bf(acc[1][t][r]);
            size_t vt_idx = (((size_t)(b * NH + h) * DKV) + dk) * S_LEN + s;
            Vt[vt_idx] = f2bf(acc[2][t][r]);
        }
    }
}

// ---------------------------------------------------------------------------
// Kernel 3: flash attention with T5 relative bias (no 1/sqrt(dk) scale).
// grid = (S/64, H, B); block = 128 (4 waves); each wave owns 16 query rows.
// ---------------------------------------------------------------------------
__global__ __launch_bounds__(128)
void flash_attn_kernel(const unsigned short* __restrict__ Qbf,
                       const unsigned short* __restrict__ Kbf,
                       const unsigned short* __restrict__ Vt,
                       const float* __restrict__ bias_tab,
                       float* __restrict__ ctx) {
    __shared__ unsigned short Plds[4][16 * LDSTR];   // per-wave P tile (16 x 64 bf16)

    const int tid  = threadIdx.x;
    const int wave = tid >> 5;
    const int lane = tid & 31;
    const int hi   = lane >> 4;
    const int lcol = lane & 15;

    const int q0 = blockIdx.x * 64;
    const int h  = blockIdx.y;
    const int b  = blockIdx.z;

    const size_t bh = (size_t)(b * NH + h);
    const unsigned short* Q = Qbf + bh * S_LEN * DKV;
    const unsigned short* K = Kbf + bh * S_LEN * DKV;
    const unsigned short* V = Vt  + bh * DKV * S_LEN;    // [dk][s]
    const float* bias_h = bias_tab + (size_t)h * NDELTA;

    const int qrow_base = q0 + wave * 16;

    // Q fragments (dk halves 0..31 / 32..63) — loop invariant, kept in VGPRs
    FragAB qf[2];
    {
        const unsigned short* qr = Q + (size_t)(qrow_base + lcol) * DKV;
#pragma unroll
        for (int ks = 0; ks < 2; ++ks) {
            qf[ks].u[0] = *(const uint4*)(qr + ks * 32 + hi * 8);
            qf[ks].u[1] = *(const uint4*)(qr + ks * 32 + hi * 8 + 16);
        }
    }

    v8f oacc[4];
#pragma unroll
    for (int d = 0; d < 4; ++d) oacc[d] = v8f_zero();
    float mrow[8], lrow[8];
#pragma unroll
    for (int r = 0; r < 8; ++r) { mrow[r] = -1e30f; lrow[r] = 0.0f; }

    for (int kblk = 0; kblk < S_LEN; kblk += 64) {
        // ----- S = Q . K^T for a 16x64 score tile -----
        v8f sacc[4];
#pragma unroll
        for (int t = 0; t < 4; ++t) sacc[t] = v8f_zero();

#pragma unroll
        for (int ks = 0; ks < 2; ++ks) {
            FragAB kf[4];   // B-matrix: (dk x keys); K row-major [s][dk] is column-contiguous
#pragma unroll
            for (int t = 0; t < 4; ++t) {
                const unsigned short* kr =
                    K + (size_t)(kblk + t * 16 + lcol) * DKV + ks * 32 + hi * 8;
                kf[t].u[0] = *(const uint4*)(kr);
                kf[t].u[1] = *(const uint4*)(kr + 16);
            }
#pragma unroll
            for (int t = 0; t < 4; ++t)
                sacc[t] = wmma_bf16(qf[ks].v, kf[t].v, sacc[t]);
        }

        // ----- add T5 bias, online softmax -----
        float tmax[8];
#pragma unroll
        for (int r = 0; r < 8; ++r) tmax[r] = -1e30f;
#pragma unroll
        for (int t = 0; t < 4; ++t) {
            int kpos = kblk + t * 16 + lcol;
#pragma unroll
            for (int r = 0; r < 8; ++r) {
                int qpos = qrow_base + r + hi * 8;
                float s = sacc[t][r] + bias_h[kpos - qpos + (S_LEN - 1)];
                sacc[t][r] = s;
                tmax[r] = fmaxf(tmax[r], s);
            }
        }
        // row-max across the 16-lane half (C-layout: row r in lanes 0-15, r+8 in 16-31)
#pragma unroll
        for (int r = 0; r < 8; ++r) {
            float v = tmax[r];
#pragma unroll
            for (int mask = 8; mask >= 1; mask >>= 1)
                v = fmaxf(v, __shfl_xor(v, mask, 32));
            tmax[r] = v;
        }
        float esc[8];
#pragma unroll
        for (int r = 0; r < 8; ++r) {
            float mnew = fmaxf(mrow[r], tmax[r]);
            esc[r] = __expf(mrow[r] - mnew);
            mrow[r] = mnew;
        }
        float rsum[8];
#pragma unroll
        for (int r = 0; r < 8; ++r) rsum[r] = 0.0f;
#pragma unroll
        for (int t = 0; t < 4; ++t) {
#pragma unroll
            for (int r = 0; r < 8; ++r) {
                float p = __expf(sacc[t][r] - mrow[r]);
                sacc[t][r] = p;
                rsum[r] += p;
            }
        }
#pragma unroll
        for (int r = 0; r < 8; ++r) {
            float v = rsum[r];
#pragma unroll
            for (int mask = 8; mask >= 1; mask >>= 1)
                v += __shfl_xor(v, mask, 32);
            lrow[r] = lrow[r] * esc[r] + v;
        }
#pragma unroll
        for (int d = 0; d < 4; ++d)
#pragma unroll
            for (int r = 0; r < 8; ++r) oacc[d][r] *= esc[r];

        // ----- P (C-layout) -> LDS bf16 (row-major) -----
#pragma unroll
        for (int t = 0; t < 4; ++t)
#pragma unroll
            for (int r = 0; r < 8; ++r)
                Plds[wave][(r + hi * 8) * LDSTR + t * 16 + lcol] = f2bf(sacc[t][r]);
        __syncthreads();

        // ----- O += P . V   (Vt rows are key-contiguous) -----
#pragma unroll
        for (int ks = 0; ks < 2; ++ks) {
            FragAB pf;   // A-matrix: 16 x 32 keys
            const unsigned short* pr = &Plds[wave][lcol * LDSTR + ks * 32 + hi * 8];
            pf.u[0] = *(const uint4*)(pr);
            pf.u[1] = *(const uint4*)(pr + 16);
            FragAB vfg[4];  // B-matrix: (keys x dk)
#pragma unroll
            for (int d = 0; d < 4; ++d) {
                const unsigned short* vr =
                    V + (size_t)(d * 16 + lcol) * S_LEN + kblk + ks * 32 + hi * 8;
                vfg[d].u[0] = *(const uint4*)(vr);
                vfg[d].u[1] = *(const uint4*)(vr + 16);
            }
#pragma unroll
            for (int d = 0; d < 4; ++d)
                oacc[d] = wmma_bf16(pf.v, vfg[d].v, oacc[d]);
        }
        __syncthreads();
    }

    // ----- epilogue: ctx[b, s, h*64 + dk] = O / l  (heads re-interleaved) -----
#pragma unroll
    for (int d = 0; d < 4; ++d) {
#pragma unroll
        for (int r = 0; r < 8; ++r) {
            int s = qrow_base + r + hi * 8;
            int dk = d * 16 + lcol;
            ctx[(size_t)(b * S_LEN + s) * DMODEL + h * DKV + dk] = oacc[d][r] / lrow[r];
        }
    }
}

// ---------------------------------------------------------------------------
// Kernel 4: output projection  out[4096,1024] = ctx(f32) . Wo(f32), via bf16 WMMA
// Block = 128 threads (4 waves). Block tile 128x64, wave tile 32x64:
// doubles B-fragment reuse per accumulator vs the 16x64 tile.
// ---------------------------------------------------------------------------
__global__ __launch_bounds__(128)
void out_proj_kernel(const float* __restrict__ A,
                     const float* __restrict__ Wo,
                     float* __restrict__ out) {
    __shared__ unsigned short Alds[128 * LDSTR];
    __shared__ unsigned short Blds[64 * LDSTR];

    const int tid  = threadIdx.x;
    const int wave = tid >> 5;
    const int lane = tid & 31;
    const int hi   = lane >> 4;
    const int lcol = lane & 15;
    const int m0 = blockIdx.y * 128;
    const int n0 = blockIdx.x * 64;

    v8f acc[2][4];
#pragma unroll
    for (int s = 0; s < 2; ++s)
#pragma unroll
        for (int t = 0; t < 4; ++t) acc[s][t] = v8f_zero();

    const int bk = tid >> 2, bq = tid & 3;

    for (int k0 = 0; k0 < DMODEL; k0 += TK) {
        if (k0 + TK < DMODEL) {
            __builtin_prefetch(A + (size_t)(m0 + tid) * DMODEL + (k0 + TK), 0, 1);
            __builtin_prefetch(Wo + (size_t)(k0 + TK + bk) * DMODEL + n0 + bq * 16, 0, 1);
        }
        // --- stage A tile (128 rows x 32 k, one row per thread) ---
        {
            const float4* src = (const float4*)(A + (size_t)(m0 + tid) * DMODEL + k0);
            us4* dst = (us4*)(Alds + tid * LDSTR);
#pragma unroll
            for (int j = 0; j < 8; ++j) {
                float4 f = src[j];
                us4 hh;
                hh.x = f2bf(f.x); hh.y = f2bf(f.y); hh.z = f2bf(f.z); hh.w = f2bf(f.w);
                dst[j] = hh;
            }
        }
        // --- stage B tile transposed ---
        {
            const float4* src = (const float4*)(Wo + (size_t)(k0 + bk) * DMODEL + n0 + bq * 16);
#pragma unroll
            for (int j = 0; j < 4; ++j) {
                float4 f = src[j];
                int nb = bq * 16 + j * 4;
                Blds[(nb + 0) * LDSTR + bk] = f2bf(f.x);
                Blds[(nb + 1) * LDSTR + bk] = f2bf(f.y);
                Blds[(nb + 2) * LDSTR + bk] = f2bf(f.z);
                Blds[(nb + 3) * LDSTR + bk] = f2bf(f.w);
            }
        }
        __syncthreads();

        FragAB af[2];
#pragma unroll
        for (int s = 0; s < 2; ++s) {
            const unsigned short* ap = Alds + (wave * 32 + s * 16 + lcol) * LDSTR + hi * 8;
            af[s].u[0] = *(const uint4*)(ap);
            af[s].u[1] = *(const uint4*)(ap + 16);
        }
        FragAB bfg[4];
#pragma unroll
        for (int t = 0; t < 4; ++t) {
            const unsigned short* bp = &Blds[(t * 16 + lcol) * LDSTR + hi * 8];
            bfg[t].u[0] = *(const uint4*)(bp);
            bfg[t].u[1] = *(const uint4*)(bp + 16);
        }
#pragma unroll
        for (int s = 0; s < 2; ++s)
#pragma unroll
            for (int t = 0; t < 4; ++t)
                acc[s][t] = wmma_bf16(af[s].v, bfg[t].v, acc[s][t]);
        __syncthreads();
    }

#pragma unroll
    for (int s = 0; s < 2; ++s) {
#pragma unroll
        for (int t = 0; t < 4; ++t) {
#pragma unroll
            for (int r = 0; r < 8; ++r) {
                int m = m0 + wave * 32 + s * 16 + r + hi * 8;
                int n = n0 + t * 16 + lcol;
                out[(size_t)m * DMODEL + n] = acc[s][t][r];
            }
        }
    }
}

// ---------------------------------------------------------------------------
// Launcher
// ---------------------------------------------------------------------------
extern "C" void kernel_launch(void* const* d_in, const int* in_sizes, int n_in,
                              void* d_out, int out_size, void* d_ws, size_t ws_size,
                              hipStream_t stream) {
    const float* X  = (const float*)d_in[0];   // hidden_states [2,2048,1024]
    const float* Wq = (const float*)d_in[1];
    const float* Wk = (const float*)d_in[2];
    const float* Wv = (const float*)d_in[3];
    const float* Wo = (const float*)d_in[4];
    const float* rb = (const float*)d_in[5];   // rel_bias [32,16]
    float* out = (float*)d_out;

    char* ws = (char*)d_ws;
    unsigned short* Qbf = (unsigned short*)(ws);                       //  8 MB  [B,H,S,dk] bf16
    unsigned short* Kbf = (unsigned short*)(ws + ((size_t)8  << 20));  //  8 MB
    unsigned short* Vt  = (unsigned short*)(ws + ((size_t)16 << 20));  //  8 MB  [B,H,dk,S] bf16
    float*          ctx = (float*)        (ws + ((size_t)24 << 20));   // 16 MB  [B,S,H*dk] f32
    float*     bias_tab = (float*)        (ws + ((size_t)40 << 20));   // 256 KB [H, 4095] f32

    bias_table_kernel<<<(NH * NDELTA + 255) / 256, 256, 0, stream>>>(rb, bias_tab);

    qkv_proj_kernel<<<dim3(DMODEL / 64, MTOT / 64), 128, 0, stream>>>(
        X, Wq, Wk, Wv, Qbf, Kbf, Vt);

    flash_attn_kernel<<<dim3(S_LEN / 64, NH, BATCH), 128, 0, stream>>>(
        Qbf, Kbf, Vt, bias_tab, ctx);

    out_proj_kernel<<<dim3(DMODEL / 64, MTOT / 128), 128, 0, stream>>>(
        ctx, Wo, out);
}
